// LevelsOfExperts_56358560858365
// MI455X (gfx1250) — matrix operand order repaired and checked by reading, compiled, not run on hardware
//
#include <hip/hip_runtime.h>
#include <hip/hip_bf16.h>
#include <cstdint>

// ---------------- problem constants ----------------
#define B_      8
#define P_      32768
#define T_      (B_ * P_)        // 262144 rows
#define LATENT  256
#define TILE_M  64
#define NWAVES  8

typedef __bf16 v16bf __attribute__((ext_vector_type(16)));
typedef float  v8f   __attribute__((ext_vector_type(8)));

union Frag16 { uint4 q[2]; v16bf v; };

// per-layer geometry for WMMA layers (0..6); layer 7 is a GEMV
__device__ __constant__ const int c_kpad[7] = {288, 512, 512, 512, 512, 512, 512};
__device__ __constant__ const int c_opad[7] = {512, 512, 512, 256, 512, 512, 512};
__device__ __constant__ const int c_outd[7] = {512, 512, 512, 253, 512, 512, 512};
__device__ __constant__ const int c_ind [7] = {259, 512, 512, 512, 512, 512, 512};
// bf16-element offsets of each layer's converted weights in d_ws
__device__ __constant__ const size_t c_woff[7] = {
    0ull, 1179648ull, 3276800ull, 5373952ull, 6422528ull, 8519680ull, 10616832ull};

__device__ __forceinline__ unsigned short f2bf(float f) {
  unsigned u = __float_as_uint(f);
  unsigned r = u + 0x7fffu + ((u >> 16) & 1u);   // round-to-nearest-even
  return (unsigned short)(r >> 16);
}
__device__ __forceinline__ float bf2f(unsigned short h) {
  return __uint_as_float(((unsigned)h) << 16);
}

// ---------------- weight conversion: fp32 -> padded bf16 in d_ws ----------------
__global__ __launch_bounds__(256) void loe_prep_weights(
    const float* W0, const float* W1, const float* W2, const float* W3,
    const float* W4, const float* W5, const float* W6, unsigned short* ws) {
  const int l = blockIdx.y;
  const float* Wl[7] = {W0, W1, W2, W3, W4, W5, W6};
  const float* W = Wl[l];
  const int kp = c_kpad[l], op = c_opad[l], od = c_outd[l], id = c_ind[l];
  const size_t total = 8ull * (size_t)op * (size_t)kp;
  for (size_t i = (size_t)blockIdx.x * blockDim.x + threadIdx.x; i < total;
       i += (size_t)gridDim.x * blockDim.x) {
    int k = (int)(i % (size_t)kp);
    size_t r = i / (size_t)kp;
    int o = (int)(r % (size_t)op);
    int e = (int)(r / (size_t)op);
    float v = 0.0f;
    if (o < od && k < id) v = W[((size_t)e * od + o) * id + k];
    ws[c_woff[l] + i] = f2bf(v);
  }
}

// ---------------- main fused kernel ----------------
// 64 rows per block, 8 waves; bf16 activations in LDS; masked-A WMMA over experts.
// Each wave owns a PAIR of N-tiles so every masked A fragment feeds two WMMAs.
__global__ __launch_bounds__(256) void loe_mlp(
    const float* lat, const float* xyz,
    const float* b0, const float* b1, const float* b2, const float* b3,
    const float* b4, const float* b5, const float* b6,
    const float* W7, const float* b7,
    const unsigned short* ws, float* out) {
  extern __shared__ unsigned char smem[];
  unsigned short* xbuf = (unsigned short*)smem;            // 64 x 288 bf16 (concat input, zero padded)
  unsigned short* act0 = xbuf + 64 * 288;                  // 64 x 512 bf16
  unsigned short* act1 = act0 + 64 * 512;                  // 64 x 512 bf16
  float* xyzs   = (float*)(act1 + 64 * 512);               // 64 x 3 fp32
  int*   tids   = (int*)(xyzs + 64 * 3);                   // 64
  int*   presentp = tids + 64;                             // 1

  const int tid  = (int)threadIdx.x;
  const int lane = tid & 31;
  const int wave = tid >> 5;
  const long t0  = (long)blockIdx.x * TILE_M;

  // ---- stage concat input x = [lat(256) | xyz(3) | 0-pad(29)] as bf16, keep xyz fp32 ----
  for (int i = tid; i < 64 * 288; i += 256) {
    int r = i / 288, c = i % 288;
    long t = t0 + r;
    int  b = (int)(t / P_);
    float v = 0.0f;
    if (c < 256)      v = lat[(size_t)b * LATENT + c];
    else if (c < 259) v = xyz[(size_t)t * 3 + (c - 256)];
    xbuf[i] = f2bf(v);
  }
  for (int i = tid; i < 192; i += 256) {
    int r = i / 3, c = i % 3;
    xyzs[i] = xyz[(size_t)(t0 + r) * 3 + c];
  }
  __syncthreads();

  const float* biases[7] = {b0, b1, b2, b3, b4, b5, b6};

  Frag16 zf;
  zf.q[0] = make_uint4(0u, 0u, 0u, 0u);
  zf.q[1] = zf.q[0];

  const int mrow  = lane & 15;   // row within 16-row tile (also N column within B/D)
  const int khalf = lane >> 4;   // lane-half selects K sub-chunks

  for (int layer = 0; layer < 7; ++layer) {
    // ---- per-row expert id for this level ----
    if (tid < 64) {
      float alpha = (float)(1 << (layer + 1));
      int tx = ((int)floorf(alpha * xyzs[tid * 3 + 0])) & 1;
      int ty = ((int)floorf(alpha * xyzs[tid * 3 + 1])) & 1;
      int tz = ((int)floorf(alpha * xyzs[tid * 3 + 2])) & 1;
      tids[tid] = tx + 2 * ty + 4 * tz;
    }
    __syncthreads();
    if (tid == 0) {
      int pm = 0;
      for (int r = 0; r < 64; ++r) pm |= 1 << tids[r];
      *presentp = pm;
    }
    __syncthreads();
    const int present = __builtin_amdgcn_readfirstlane(*presentp);

    const unsigned short* in  = (layer == 0) ? xbuf : ((layer & 1) ? act0 : act1);
    unsigned short*       ob  = (layer & 1) ? act1 : act0;
    const int instride = (layer == 0) ? 288 : 512;
    const int ksteps   = instride / 32;
    const int kp = c_kpad[layer], op = c_opad[layer], od = c_outd[layer];
    const unsigned short* W  = ws + c_woff[layer];
    const float*        bias = biases[layer];
    const int NT = op / 16;

    int mytid[4];
#pragma unroll
    for (int mt = 0; mt < 4; ++mt) mytid[mt] = tids[mt * 16 + mrow];

    // pair of adjacent N-tiles per wave iteration
    for (int nt0 = wave * 2; nt0 < NT; nt0 += 2 * NWAVES) {
      v8f acc[2][4];
#pragma unroll
      for (int j = 0; j < 2; ++j)
#pragma unroll
        for (int mt = 0; mt < 4; ++mt)
          acc[j][mt] = (v8f){0.f, 0.f, 0.f, 0.f, 0.f, 0.f, 0.f, 0.f};

      // base of this wave's B rows (two 16-col tiles are contiguous rows of W^T)
      const unsigned short* Wn = W + (size_t)(nt0 * 16 + mrow) * kp + khalf * 8;

      for (int ks = 0; ks < ksteps; ++ks) {
        const int kb = ks * 32;
        // A fragments: 16x32 bf16, ISA layout (two 16B chunks per lane)
        Frag16 a[4];
#pragma unroll
        for (int mt = 0; mt < 4; ++mt) {
          const unsigned short* rp = in + (size_t)(mt * 16 + mrow) * instride + kb + khalf * 8;
          a[mt].q[0] = *(const uint4*)(rp);
          a[mt].q[1] = *(const uint4*)(rp + 16);
        }
        for (int e = 0; e < 8; ++e) {
          if (!((present >> e) & 1)) continue;  // scalar-uniform skip
          // B fragments for both N-tiles: 32x16 bf16 of W[e]^T
          const unsigned short* bp = Wn + (size_t)e * op * kp + kb;
          Frag16 bfr[2];
#pragma unroll
          for (int j = 0; j < 2; ++j) {
            const unsigned short* bj = bp + (size_t)j * 16 * kp;
            bfr[j].q[0] = *(const uint4*)(bj);
            bfr[j].q[1] = *(const uint4*)(bj + 16);
          }
          // pull next K-chunk of this expert's block toward the WGP (global_prefetch_b8)
          if (ks + 1 < ksteps) __builtin_prefetch(bp + 32, 0, 0);

#pragma unroll
          for (int mt = 0; mt < 4; ++mt) {
            v16bf am = (mytid[mt] == e) ? a[mt].v : zf.v;   // mask rows not owned by expert e
            acc[0][mt] = __builtin_amdgcn_wmma_f32_16x16x32_bf16(
                false, am, false, bfr[0].v, (short)0, acc[0][mt], false, false);
            acc[1][mt] = __builtin_amdgcn_wmma_f32_16x16x32_bf16(
                false, am, false, bfr[1].v, (short)0, acc[1][mt], false, false);
          }
        }
      }
      // ---- epilogue: bias + ReLU, bf16 store to ping-pong LDS ----
#pragma unroll
      for (int j = 0; j < 2; ++j) {
        const int n = (nt0 + j) * 16 + mrow;
        const float bv = (n < od) ? bias[n] : 0.0f;
#pragma unroll
        for (int mt = 0; mt < 4; ++mt) {
#pragma unroll
          for (int r = 0; r < 8; ++r) {
            int m = mt * 16 + r + khalf * 8;    // C/D layout: lane-half adds 8 to M
            float v = acc[j][mt][r] + bv;
            v = v > 0.0f ? v : 0.0f;            // ReLU (layers 0..6)
            if (n < od) ob[(size_t)m * 512 + n] = f2bf(v);
          }
        }
      }
    }
    __syncthreads();
    if (layer == 3) {  // concat: [relu(out3)(253) | x(259)] -> 512
      for (int i = tid; i < 64 * 259; i += 256) {
        int r = i / 259, c = i % 259;
        ob[(size_t)r * 512 + 253 + c] = xbuf[r * 288 + c];
      }
      __syncthreads();
    }
  }

  // ---- layer 7: out_d = 1 GEMV (fp32 weights), 4 lanes per row ----
  {
    const unsigned short* in = act0;   // layer 6 wrote act0
    const int r = tid >> 2, part = tid & 3;
    const float alpha = 256.0f;        // 2^(7+1)
    int tx = ((int)floorf(alpha * xyzs[r * 3 + 0])) & 1;
    int ty = ((int)floorf(alpha * xyzs[r * 3 + 1])) & 1;
    int tz = ((int)floorf(alpha * xyzs[r * 3 + 2])) & 1;
    int t7 = tx + 2 * ty + 4 * tz;
    const float* w = W7 + (size_t)t7 * 512;
    const unsigned short* inr = in + (size_t)r * 512;
    float s = 0.0f;
    for (int k = part * 128; k < part * 128 + 128; ++k) s += bf2f(inr[k]) * w[k];
    s += __shfl_xor(s, 1);
    s += __shfl_xor(s, 2);
    if (part == 0) out[t0 + r] = s + b7[0];
  }
}

// ---------------- host launcher ----------------
extern "C" void kernel_launch(void* const* d_in, const int* in_sizes, int n_in,
                              void* d_out, int out_size, void* d_ws, size_t ws_size,
                              hipStream_t stream) {
  (void)in_sizes; (void)n_in; (void)out_size; (void)ws_size;
  const float* lat = (const float*)d_in[0];
  const float* xyz = (const float*)d_in[1];
  const float* Ws[8];
  const float* bs[8];
  for (int i = 0; i < 8; ++i) Ws[i] = (const float*)d_in[2 + i];
  for (int i = 0; i < 8; ++i) bs[i] = (const float*)d_in[10 + i];
  unsigned short* ws = (unsigned short*)d_ws;
  float* out = (float*)d_out;

  // 1) convert weights fp32 -> padded bf16 into workspace (deterministic each call)
  dim3 gprep(512, 7, 1);
  loe_prep_weights<<<gprep, 256, 0, stream>>>(Ws[0], Ws[1], Ws[2], Ws[3], Ws[4], Ws[5], Ws[6], ws);

  // 2) fused expert-MLP
  constexpr size_t shmem = 64 * 288 * 2      // xbuf
                         + 2 * 64 * 512 * 2  // act ping-pong
                         + 64 * 3 * 4        // xyz fp32
                         + 64 * 4            // tids
                         + 16;               // present (+pad)
  loe_mlp<<<T_ / TILE_M, 256, shmem, stream>>>(
      lat, xyz, bs[0], bs[1], bs[2], bs[3], bs[4], bs[5], bs[6],
      Ws[7], bs[7], ws, out);
}
